// ActionVQVAE_49452253446164
// MI455X (gfx1250) — matrix operand order repaired and checked by reading, not compiled, not run on hardware
//
#include <hip/hip_runtime.h>
#include <hip/hip_bf16.h>
#include <math.h>

typedef __attribute__((ext_vector_type(16))) _Float16 v16h;
typedef __attribute__((ext_vector_type(8)))  _Float16 v8h;
typedef __attribute__((ext_vector_type(8)))  float    v8f;
typedef __attribute__((ext_vector_type(4)))  unsigned int u32x4;
typedef __attribute__((ext_vector_type(8)))  int i32x8;
typedef __attribute__((ext_vector_type(4)))  int i32x4;

#define B_TOT 32768
#define A_DIM 16
#define H_DIM 256
#define D_DIM 128
#define K_CB  2048
#define BETA  0.25f

#define MT       64    // batch rows per block
#define THREADS  128   // 4 waves, wave w owns rows [w*16, w*16+16)
#define LDSTR    272   // f16 elements per row in activation buffers (32B-aligned rows)
#define CHUNK    128   // codebook codes per distance iteration (32KB f16)

#if defined(__has_builtin)
#if __has_builtin(__builtin_amdgcn_tensor_load_to_lds)
#define HAVE_TDM 1
#endif
#endif
#ifndef HAVE_TDM
#define HAVE_TDM 0
#endif

// ---------- TDM: 1-D contiguous copy global(f16) -> LDS via Tensor Data Mover ----------
// D# layout per CDNA5 ISA ch.8: group0 {count=1, lds_addr, global_addr, type=2},
// group1 {data_size=1(2B), tensor_dim0, tensor_dim1=1, tile_dim0, stride0}, groups2/3 zero.
#if HAVE_TDM
__device__ __forceinline__ void tdm_copy_f16(const _Float16* gsrc, void* ldst,
                                             unsigned int nelem, unsigned int tensor_elems) {
  unsigned long long ga = (unsigned long long)(__SIZE_TYPE__)gsrc;
  unsigned int la = (unsigned int)(__SIZE_TYPE__)ldst;   // low 32 bits = LDS byte offset
  u32x4 g0;
  g0[0] = 1u;                                            // count=1, no gather/restore
  g0[1] = la;                                            // lds_addr
  g0[2] = (unsigned int)(ga & 0xffffffffu);              // global_addr[31:0]
  g0[3] = (unsigned int)((ga >> 32) & 0x01ffffffu) | (2u << 30);  // addr[56:32] | type=2
  i32x8 g1;
  g1[0] = (1 << 16);                                     // workgroup_mask=0, data_size=1 (2B)
  g1[1] = (int)((tensor_elems & 0xffffu) << 16);         // tensor_dim0[15:0]
  g1[2] = (int)((tensor_elems >> 16) & 0xffffu) | (1 << 16); // tensor_dim0[31:16], tensor_dim1=1
  g1[3] = (int)((nelem & 0xffffu) << 16);                // tile_dim0
  g1[4] = 0;                                             // tile_dim1=0, tile_dim2=0
  g1[5] = (int)tensor_elems;                             // tensor_dim0_stride[31:0]
  g1[6] = 0;
  g1[7] = 0;
  i32x4 z4 = {0, 0, 0, 0};
#if __has_include(<hip/amd_detail/amd_gfx1250_TDM.h>)
  i32x8 z8 = {0, 0, 0, 0, 0, 0, 0, 0};
  __builtin_amdgcn_tensor_load_to_lds(g0, g1, z4, z4, z8, 0);
#else
  __builtin_amdgcn_tensor_load_to_lds(g0, g1, z4, z4, 0);
#endif
}
#endif

// ---------- WMMA fragment loads (per ISA 16-bit A 16x32 / B 32x16 layouts) ----------
__device__ __forceinline__ v16h load_a_frag(const _Float16* src, int ld, int m0, int kc, int lane) {
  int m = m0 + (lane & 15);
  int h = lane >> 4;
  const _Float16* p = src + m * ld + kc + h * 8;
  v8h lo = *(const v8h*)(p);        // K = h*8 .. h*8+7
  v8h hi = *(const v8h*)(p + 16);   // K = 16 + h*8 .. +7
  v16h a;
#pragma unroll
  for (int i = 0; i < 8; ++i) { a[i] = lo[i]; a[i + 8] = hi[i]; }
  return a;
}

// W stored row-major [N][K] (x @ W.T): B[k][n] = W[n][k]; per lane K = kh*8..+7 and 16+kh*8..+7.
__device__ __forceinline__ v16h load_b_frag(const _Float16* w, int ld, int n0, int kc, int lane) {
  int n  = n0 + (lane & 15);
  int kh = lane >> 4;
  const _Float16* p = w + n * ld + kc + kh * 8;
  v8h lo = *(const v8h*)(p);
  v8h hi = *(const v8h*)(p + 16);
  v16h b;
#pragma unroll
  for (int i = 0; i < 8; ++i) { b[i] = lo[i]; b[i + 8] = hi[i]; }
  return b;
}

// ---------- fused MLP layer: dst = act(src @ W.T + b), all activations in LDS ----------
template<int NOUT, int KPAD, int KREAL, int ACT>  // ACT: 0 none, 1 relu, 2 tanh
__device__ __forceinline__ void mlp_layer(const float* __restrict__ W,
                                          const float* __restrict__ bias,
                                          const _Float16* src, _Float16* dst,
                                          _Float16* sWs, int tid, int lane, int m0) {
  constexpr int NT = NOUT / 16;
  v8f acc[NT];
#pragma unroll
  for (int nt = 0; nt < NT; ++nt)
#pragma unroll
    for (int r = 0; r < 8; ++r) acc[nt][r] = 0.0f;

  for (int kc = 0; kc < KPAD; kc += 32) {
    for (int i = tid; i < NOUT * 32; i += THREADS) {
      int n = i >> 5, k = i & 31, kk = kc + k;
      float w = (kk < KREAL) ? W[n * KREAL + kk] : 0.0f;
      sWs[n * 32 + k] = (_Float16)w;
    }
    __syncthreads();
    v16h a = load_a_frag(src, LDSTR, m0, kc, lane);
#pragma unroll
    for (int nt = 0; nt < NT; ++nt) {
      v16h b = load_b_frag(sWs, 32, nt * 16, 0, lane);
      acc[nt] = __builtin_amdgcn_wmma_f32_16x16x32_f16(false, a, false, b,
                                                       (short)0, acc[nt], false, false);
    }
    __syncthreads();
  }
  int nc = lane & 15, mg = lane >> 4;
#pragma unroll
  for (int nt = 0; nt < NT; ++nt) {
    float bv = bias[nt * 16 + nc];
#pragma unroll
    for (int r = 0; r < 8; ++r) {
      float v = acc[nt][r] + bv;
      if (ACT == 1) v = fmaxf(v, 0.0f);
      else if (ACT == 2) v = tanhf(v);
      dst[(m0 + mg * 8 + r) * LDSTR + nt * 16 + nc] = (_Float16)v;
    }
  }
  __syncthreads();
}

// ---------- fused VQ-VAE forward ----------
__global__ __launch_bounds__(THREADS)
void vqvae_fused(const float* __restrict__ action,
                 const float* __restrict__ We1, const float* __restrict__ be1,
                 const float* __restrict__ We2, const float* __restrict__ be2,
                 const float* __restrict__ We3, const float* __restrict__ be3,
                 const float* __restrict__ E,
                 const float* __restrict__ Wd1, const float* __restrict__ bd1,
                 const float* __restrict__ Wd2, const float* __restrict__ bd2,
                 const float* __restrict__ Wh,  const float* __restrict__ bh,
                 float* __restrict__ accum,
                 const float* __restrict__ e2g,      // ||E_k||^2 precomputed (TDM path)
                 const _Float16* __restrict__ Ef16,  // f16 copy of E (TDM path)
                 int use_tdm) {
  __shared__ __align__(32) _Float16 sBufA[MT * LDSTR];   // 34816 B
  __shared__ __align__(32) _Float16 sBufB[MT * LDSTR];   // 34816 B
  __shared__ __align__(32) unsigned char sScr[67584];    // E double-buffer / W slices / reductions

  _Float16* sWs   = (_Float16*)sScr;            // W slices (<=16KB)
  _Float16* sE0   = (_Float16*)sScr;            // E chunk ping (32KB)
  _Float16* sE1   = (_Float16*)(sScr + 32768);  // E chunk pong (32KB)
  float*    sE2f  = (float*)(sScr + 65536);     // fallback ||E||^2 (512B)
  float*    sRedV = (float*)sScr;               // argmin reduce vals (4KB)  [after dist loop]
  int*      sRedI = (int*)(sScr + 4096);        // argmin reduce idxs (4KB)
  int*      sIdx  = (int*)(sScr + 8192);        // winning code per row (256B)
  float*    sLoss = (float*)(sScr + 66048);     // loss partials (1KB)

  const int tid  = threadIdx.x;
  const int lane = tid & 31;
  const int m0   = (tid >> 5) * 16;
  const int g0   = blockIdx.x * MT;

  // ---- stage action tile (K=16 zero-padded to 32) ----
  for (int i = tid; i < MT * 32; i += THREADS) {
    int m = i >> 5, k = i & 31;
    float v = (k < A_DIM) ? action[(size_t)(g0 + m) * A_DIM + k] : 0.0f;
    sBufA[m * LDSTR + k] = (_Float16)v;
  }
  __syncthreads();

  // ---- encoder ----
  mlp_layer<H_DIM, 32,  A_DIM, 1>(We1, be1, sBufA, sBufB, sWs, tid, lane, m0);
  mlp_layer<H_DIM, 256, H_DIM, 1>(We2, be2, sBufB, sBufA, sWs, tid, lane, m0);
  mlp_layer<D_DIM, 256, H_DIM, 0>(We3, be3, sBufA, sBufB, sWs, tid, lane, m0); // enc -> sBufB

  // ---- codebook search: argmin_k (||E_k||^2 - 2 enc.E_k) via WMMA ----
  v16h ae[4];
#pragma unroll
  for (int t = 0; t < 4; ++t) ae[t] = load_a_frag(sBufB, LDSTR, m0, t * 32, lane);

  float minV[8]; int minI[8];
#pragma unroll
  for (int r = 0; r < 8; ++r) { minV[r] = 3.4e38f; minI[r] = 0; }

  constexpr int NCH = K_CB / CHUNK;   // 16 chunks

#if HAVE_TDM
  if (use_tdm && (tid >> 5) == 0)     // wave 0 primes the pipeline via the Tensor Data Mover
    tdm_copy_f16(Ef16, sE0, CHUNK * D_DIM, K_CB * D_DIM);
#endif

  for (int ch = 0; ch < NCH; ++ch) {
    const int cb = ch * CHUNK;
    const _Float16* cur = sE0;
#if HAVE_TDM
    if (use_tdm) {
      cur = (ch & 1) ? sE1 : sE0;
      if ((tid >> 5) == 0) {
        if (ch + 1 < NCH) {           // stream next chunk while we compute on this one
          tdm_copy_f16(Ef16 + (size_t)(cb + CHUNK) * D_DIM,
                       (ch & 1) ? (void*)sE0 : (void*)sE1, CHUNK * D_DIM, K_CB * D_DIM);
          __builtin_amdgcn_s_wait_tensorcnt(1);   // current chunk complete, next in flight
        } else {
          __builtin_amdgcn_s_wait_tensorcnt(0);
        }
      }
      __syncthreads();
    } else
#endif
    { // fallback: synchronous f32 staging + on-the-fly ||E||^2
      const float* er = E + (size_t)(cb + tid) * D_DIM;
      if (cb + CHUNK < K_CB)
        __builtin_prefetch(E + (size_t)(cb + CHUNK + tid) * D_DIM, 0, 1);
      float ss = 0.0f;
#pragma unroll 4
      for (int d = 0; d < D_DIM; d += 4) {
        float4 v = *(const float4*)(er + d);
        ss += v.x * v.x + v.y * v.y + v.z * v.z + v.w * v.w;
        sE0[tid * D_DIM + d + 0] = (_Float16)v.x;
        sE0[tid * D_DIM + d + 1] = (_Float16)v.y;
        sE0[tid * D_DIM + d + 2] = (_Float16)v.z;
        sE0[tid * D_DIM + d + 3] = (_Float16)v.w;
      }
      sE2f[tid] = ss;
      __syncthreads();
    }

#pragma unroll
    for (int nt = 0; nt < CHUNK / 16; ++nt) {
      v8f acc;
#pragma unroll
      for (int r = 0; r < 8; ++r) acc[r] = 0.0f;
#pragma unroll
      for (int t = 0; t < 4; ++t) {
        v16h b = load_b_frag(cur, D_DIM, nt * 16, t * 32, lane);
        acc = __builtin_amdgcn_wmma_f32_16x16x32_f16(false, ae[t], false, b,
                                                     (short)0, acc, false, false);
      }
      int c = nt * 16 + (lane & 15);
      int kidx = cb + c;
      float e2 = use_tdm ? e2g[kidx] : sE2f[c];
#pragma unroll
      for (int r = 0; r < 8; ++r) {
        float val = e2 - 2.0f * acc[r];
        if (val < minV[r]) { minV[r] = val; minI[r] = kidx; }
      }
    }
    __syncthreads();
  }

  // ---- cross-lane argmin reduce ----
  {
    int nc = lane & 15, mg = lane >> 4;
#pragma unroll
    for (int r = 0; r < 8; ++r) {
      int m = m0 + mg * 8 + r;
      sRedV[m * 16 + nc] = minV[r];
      sRedI[m * 16 + nc] = minI[r];
    }
  }
  __syncthreads();
  if (tid < MT) {
    float bv = sRedV[tid * 16]; int bi = sRedI[tid * 16];
    for (int n = 1; n < 16; ++n) {
      float v = sRedV[tid * 16 + n]; int ii = sRedI[tid * 16 + n];
      if (v < bv || (v == bv && ii < bi)) { bv = v; bi = ii; }
    }
    sIdx[tid] = bi;
  }
  __syncthreads();

  // ---- gather q, VQ-loss partial ((1+beta)*mean((enc-q)^2); q_st==q in fwd) ----
  float vq_part = 0.0f;
  for (int i = tid; i < MT * D_DIM; i += THREADS) {
    int m = i >> 7, d = i & 127;
    float qv = E[(size_t)sIdx[m] * D_DIM + d];
    float ev = (float)sBufB[m * LDSTR + d];
    float df = ev - qv;
    vq_part += df * df;
    sBufA[m * LDSTR + d] = (_Float16)qv;
  }
  __syncthreads();

  // ---- decoder ----
  mlp_layer<H_DIM, 128, D_DIM, 1>(Wd1, bd1, sBufA, sBufB, sWs, tid, lane, m0);
  mlp_layer<H_DIM, 256, H_DIM, 1>(Wd2, bd2, sBufB, sBufA, sWs, tid, lane, m0);
  mlp_layer<A_DIM, 256, H_DIM, 2>(Wh,  bh,  sBufA, sBufB, sWs, tid, lane, m0); // recons -> sBufB

  // ---- reconstruction loss partial ----
  float rec_part = 0.0f;
  for (int i = tid; i < MT * A_DIM; i += THREADS) {
    int m = i >> 4, d = i & 15;
    float r = (float)sBufB[m * LDSTR + d];
    float a = action[(size_t)(g0 + m) * A_DIM + d];
    float df = r - a;
    rec_part += df * df;
  }

  sLoss[tid] = rec_part;
  sLoss[THREADS + tid] = vq_part;
  __syncthreads();
  if (tid == 0) {
    float rs = 0.0f, vs = 0.0f;
    for (int i = 0; i < THREADS; ++i) { rs += sLoss[i]; vs += sLoss[THREADS + i]; }
    atomicAdd(&accum[0], rs);
    atomicAdd(&accum[1], vs);
  }
}

// ---------- prep: E -> f16 copy + ||E_k||^2 (enables TDM byte-copy streaming) ----------
__global__ void vq_prep_E(const float* __restrict__ E, _Float16* __restrict__ Ef16,
                          float* __restrict__ e2g) {
  int k = blockIdx.x * blockDim.x + threadIdx.x;   // one code row per thread
  if (k >= K_CB) return;
  const float* er = E + (size_t)k * D_DIM;
  _Float16* fr = Ef16 + (size_t)k * D_DIM;
  float ss = 0.0f;
#pragma unroll 4
  for (int d = 0; d < D_DIM; d += 4) {
    float4 v = *(const float4*)(er + d);
    ss += v.x * v.x + v.y * v.y + v.z * v.z + v.w * v.w;
    fr[d + 0] = (_Float16)v.x; fr[d + 1] = (_Float16)v.y;
    fr[d + 2] = (_Float16)v.z; fr[d + 3] = (_Float16)v.w;
  }
  e2g[k] = ss;
}

__global__ void vq_init(float* accum) { accum[0] = 0.0f; accum[1] = 0.0f; }

__global__ void vq_finalize(const float* __restrict__ accum, float* __restrict__ out) {
  float recons = accum[0] / (float)(B_TOT * A_DIM);
  float vq     = (1.0f + BETA) * (accum[1] / (float)(B_TOT * D_DIM));
  out[0] = recons + vq;
}

extern "C" void kernel_launch(void* const* d_in, const int* in_sizes, int n_in,
                              void* d_out, int out_size, void* d_ws, size_t ws_size,
                              hipStream_t stream) {
  const float* action = (const float*)d_in[0];
  const float* We1 = (const float*)d_in[1];  const float* be1 = (const float*)d_in[2];
  const float* We2 = (const float*)d_in[3];  const float* be2 = (const float*)d_in[4];
  const float* We3 = (const float*)d_in[5];  const float* be3 = (const float*)d_in[6];
  const float* E   = (const float*)d_in[7];
  const float* Wd1 = (const float*)d_in[8];  const float* bd1 = (const float*)d_in[9];
  const float* Wd2 = (const float*)d_in[10]; const float* bd2 = (const float*)d_in[11];
  const float* Wh  = (const float*)d_in[12]; const float* bh  = (const float*)d_in[13];

  // workspace: [0,8) accum | [256, 8448) e2 | [8448, 8448+512K) E in f16
  float*     accum = (float*)d_ws;
  float*     e2g   = (float*)((char*)d_ws + 256);
  _Float16*  Ef16  = (_Float16*)((char*)d_ws + 8448);
  const size_t need = 8448 + (size_t)K_CB * D_DIM * sizeof(_Float16);
  int use_tdm = (HAVE_TDM && ws_size >= need) ? 1 : 0;

  vq_init<<<1, 1, 0, stream>>>(accum);
  if (use_tdm)
    vq_prep_E<<<K_CB / 128, 128, 0, stream>>>(E, Ef16, e2g);
  vqvae_fused<<<B_TOT / MT, THREADS, 0, stream>>>(action, We1, be1, We2, be2, We3, be3,
                                                  E, Wd1, bd1, Wd2, bd2, Wh, bh,
                                                  accum, e2g, Ef16, use_tdm);
  vq_finalize<<<1, 1, 0, stream>>>(accum, (float*)d_out);
}
